// generator_gnn_38302518345769
// MI455X (gfx1250) — 8-bit matrix operand order repaired and checked by reading, compiled, not run on hardware
//
#include <hip/hip_runtime.h>
#include <hip/hip_bf16.h>
#include <stdint.h>

#define N_NODES 2048
#define LATENT  128
#define EMB_DIM 8
#define BATCH   64
#define KCHUNKS 4
#define KCH     (N_NODES / KCHUNKS)

typedef int   v8i __attribute__((ext_vector_type(8)));
typedef float v8f __attribute__((ext_vector_type(8)));
typedef float v2f __attribute__((ext_vector_type(2)));

__global__ void build_a_u8(const float* __restrict__ adj, uint8_t* __restrict__ a) {
    int idx = blockIdx.x * blockDim.x + threadIdx.x;
    const int total = N_NODES * N_NODES;
    for (; idx < total; idx += gridDim.x * blockDim.x) {
        int n = idx >> 11, m = idx & (N_NODES - 1);
        a[idx] = (adj[idx] != 0.0f && n != m) ? 1 : 0;
    }
}

__global__ void row_deg_inv(const uint8_t* __restrict__ a, float* __restrict__ dinv) {
    __shared__ float red[256];
    int row = blockIdx.x;
    const uint8_t* r = a + (size_t)row * N_NODES;
    int t = threadIdx.x;
    float s = 0.f;
    for (int c = t; c < N_NODES; c += 256) s += (float)r[c];
    red[t] = s;
    __syncthreads();
    for (int off = 128; off > 0; off >>= 1) {
        if (t < off) red[t] += red[t + off];
        __syncthreads();
    }
    if (t == 0) {
        float d = red[0];
        dinv[row] = (d > 0.f) ? rsqrtf(d) : 0.f;
    }
}

__global__ void __launch_bounds__(128, 1)
twohop_wmma(const uint8_t* __restrict__ a, uint8_t* __restrict__ th) {
    int wid  = (blockIdx.x * blockDim.x + threadIdx.x) >> 5;
    int lane = threadIdx.x & 31;
    int bm = wid >> 6;
    int bn = wid & 63;
    int half = lane >> 4;
    int m16  = lane & 15;
    const uint8_t* arow0 = a + (size_t)(bm * 64 + m16) * N_NODES;
    const uint8_t* bcol0 = a + (size_t)(bn * 32 + m16) * N_NODES;
    v8i acc[4][2] = {};
    for (int k0 = 0; k0 < N_NODES; k0 += 64) {
        v8i af[4];
        #pragma unroll
        for (int r = 0; r < 4; ++r) {
            const uint8_t* ar = arow0 + (size_t)(r * 16) * N_NODES;
            #pragma unroll
            for (int v = 0; v < 8; ++v) {
                int ka = k0 + (v >> 1) * 16 + half * 8 + (v & 1) * 4;
                af[r][v] = *(const int*)(ar + ka);
            }
        }
        #pragma unroll
        for (int c = 0; c < 2; ++c) {
            const uint8_t* bc = bcol0 + (size_t)(c * 16) * N_NODES;
            v8i bf;
            #pragma unroll
            for (int v = 0; v < 8; ++v) {
                int kb = k0 + (v >> 1) * 16 + half * 8 + (v & 1) * 4;
                bf[v] = *(const int*)(bc + kb);
            }
            #pragma unroll
            for (int r = 0; r < 4; ++r)
                acc[r][c] = __builtin_amdgcn_wmma_i32_16x16x64_iu8(false, af[r], false, bf,
                                                                  acc[r][c], false, false);
        }
    }
    #pragma unroll
    for (int r = 0; r < 4; ++r) {
        #pragma unroll
        for (int c = 0; c < 2; ++c) {
            #pragma unroll
            for (int j = 0; j < 8; ++j) {
                int rr = bm * 64 + r * 16 + half * 8 + j;
                int cc = bn * 32 + c * 16 + m16;
                size_t idx = (size_t)rr * N_NODES + cc;
                int av = (int)a[idx];
                int pred = (int)(acc[r][c][j] > 0) & (int)(av == 0) & (int)(rr != cc);
                th[idx] = (uint8_t)pred;
            }
        }
    }
}

__global__ void build_a1a2(const uint8_t* __restrict__ a, const uint8_t* __restrict__ th,
                           const float* __restrict__ d1, const float* __restrict__ d2,
                           float* __restrict__ a1, float* __restrict__ a2) {
    int idx = blockIdx.x * blockDim.x + threadIdx.x;
    const int total = N_NODES * N_NODES;
    for (; idx < total; idx += gridDim.x * blockDim.x) {
        int n = idx >> 11, m = idx & (N_NODES - 1);
        a1[idx] = a[idx]  ? d1[n] * d1[m] : 0.f;
        a2[idx] = th[idx] ? d2[n] * d2[m] : 0.f;
    }
}

__global__ void compute_st(const float* __restrict__ x, const int* __restrict__ cat,
                           const float* __restrict__ emb, const float* __restrict__ w_map,
                           const float* __restrict__ b_map, float* __restrict__ st) {
    int n = blockIdx.x;
    int b = threadIdx.x;
    const float* xb = x + b * LATENT;
    const float* eb = emb + cat[b] * EMB_DIM;
    float acc = b_map[n];
    for (int f = 0; f < LATENT; ++f)  acc += xb[f] * w_map[f * N_NODES + n];
    for (int f = 0; f < EMB_DIM; ++f) acc += eb[f] * w_map[(LATENT + f) * N_NODES + n];
    st[n * BATCH + b] = acc;
}

__global__ void __launch_bounds__(128, 1)
gemm_nt_wmma_f32(const float* __restrict__ M, const float* __restrict__ X,
                 float* __restrict__ Ypart) {
    int wave = threadIdx.x >> 5;
    int lane = threadIdx.x & 31;
    int tm = blockIdx.x * 4 + wave;
    int kc = blockIdx.y;
    int half = lane >> 4;
    int m16  = lane & 15;
    const float* Mr = M + (size_t)(tm * 16 + m16) * N_NODES;
    v8f c0 = {}, c1 = {}, c2 = {}, c3 = {};
    for (int k0 = kc * KCH; k0 < (kc + 1) * KCH; k0 += 4) {
        v2f af;
        af[0] = Mr[k0 + half * 2 + 0];
        af[1] = Mr[k0 + half * 2 + 1];
        const float* X0 = X + (size_t)(k0 + half * 2) * BATCH + m16;
        v2f b0, b1, b2, b3;
        b0[0] = X0[0];  b0[1] = X0[BATCH];
        b1[0] = X0[16]; b1[1] = X0[BATCH + 16];
        b2[0] = X0[32]; b2[1] = X0[BATCH + 32];
        b3[0] = X0[48]; b3[1] = X0[BATCH + 48];
        c0 = __builtin_amdgcn_wmma_f32_16x16x4_f32(false, af, false, b0, (short)0, c0, false, false);
        c1 = __builtin_amdgcn_wmma_f32_16x16x4_f32(false, af, false, b1, (short)0, c1, false, false);
        c2 = __builtin_amdgcn_wmma_f32_16x16x4_f32(false, af, false, b2, (short)0, c2, false, false);
        c3 = __builtin_amdgcn_wmma_f32_16x16x4_f32(false, af, false, b3, (short)0, c3, false, false);
    }
    float* Yp = Ypart + (size_t)kc * N_NODES * BATCH;
    #pragma unroll
    for (int j = 0; j < 8; ++j) {
        float* Yr = Yp + (size_t)(tm * 16 + half * 8 + j) * BATCH + m16;
        Yr[0]  = c0[j];
        Yr[16] = c1[j];
        Yr[32] = c2[j];
        Yr[48] = c3[j];
    }
}

__global__ void reduce_kchunks(const float* __restrict__ Ypart, float* __restrict__ Y) {
    int idx = blockIdx.x * blockDim.x + threadIdx.x;
    if (idx >= N_NODES * BATCH) return;
    const size_t S = (size_t)N_NODES * BATCH;
    Y[idx] = Ypart[idx] + Ypart[idx + S] + Ypart[idx + 2 * S] + Ypart[idx + 3 * S];
}

__global__ void combine_out(const float* __restrict__ st, const float* __restrict__ t1,
                            const float* __restrict__ t2, const float* __restrict__ u11,
                            const float* __restrict__ u12, const float* __restrict__ u21,
                            const float* __restrict__ u22, const float* __restrict__ w_embed,
                            const float* __restrict__ w_out, float* __restrict__ out) {
    int idx = blockIdx.x * blockDim.x + threadIdx.x;
    if (idx >= BATCH * N_NODES) return;
    int b = idx >> 11;
    int n = idx & (N_NODES - 1);
    float c[7];
    #pragma unroll
    for (int g = 0; g < 7; ++g) {
        float s = 0.f;
        #pragma unroll
        for (int h = 0; h < 8; ++h) s += w_embed[h] * w_out[g * 8 + h];
        c[g] = s;
    }
    size_t p = (size_t)n * BATCH + b;
    out[idx] = c[0] * st[p] + c[1] * t1[p] + c[2] * t2[p] +
               c[3] * u11[p] + c[4] * u12[p] + c[5] * u21[p] + c[6] * u22[p];
}

extern "C" void kernel_launch(void* const* d_in, const int* in_sizes, int n_in,
                              void* d_out, int out_size, void* d_ws, size_t ws_size,
                              hipStream_t stream) {
    const float* x    = (const float*)d_in[0];
    const int*   cat  = (const int*)d_in[1];
    const float* adj  = (const float*)d_in[2];
    const float* emb  = (const float*)d_in[3];
    const float* wmap = (const float*)d_in[4];
    const float* bmap = (const float*)d_in[5];
    const float* wemb = (const float*)d_in[6];
    const float* wout = (const float*)d_in[7];
    float* out = (float*)d_out;

    char* ws = (char*)d_ws;
    const size_t NN = (size_t)N_NODES * N_NODES;
    uint8_t* a_u8  = (uint8_t*)(ws);
    uint8_t* th_u8 = (uint8_t*)(ws + NN);
    float*   a1    = (float*)(ws + 2 * NN);
    float*   a2    = (float*)(ws + 2 * NN + 4 * NN);
    char* p = ws + 10 * NN;
    const size_t TB = (size_t)N_NODES * BATCH * sizeof(float);
    float* st  = (float*)p; p += TB;
    float* t1  = (float*)p; p += TB;
    float* t2  = (float*)p; p += TB;
    float* u11 = (float*)p; p += TB;
    float* u12 = (float*)p; p += TB;
    float* u21 = (float*)p; p += TB;
    float* u22 = (float*)p; p += TB;
    float* yp  = (float*)p; p += KCHUNKS * TB;
    float* d1  = (float*)p; p += N_NODES * sizeof(float);
    float* d2  = (float*)p;

    build_a_u8<<<4096, 256, 0, stream>>>(adj, a_u8);
    row_deg_inv<<<N_NODES, 256, 0, stream>>>(a_u8, d1);
    twohop_wmma<<<512, 128, 0, stream>>>(a_u8, th_u8);
    row_deg_inv<<<N_NODES, 256, 0, stream>>>(th_u8, d2);
    build_a1a2<<<4096, 256, 0, stream>>>(a_u8, th_u8, d1, d2, a1, a2);
    compute_st<<<N_NODES, BATCH, 0, stream>>>(x, cat, emb, wmap, bmap, st);

    dim3 ggrid(32, KCHUNKS);
    gemm_nt_wmma_f32<<<ggrid, 128, 0, stream>>>(a1, st, yp);
    reduce_kchunks<<<512, 256, 0, stream>>>(yp, t1);
    gemm_nt_wmma_f32<<<ggrid, 128, 0, stream>>>(a2, st, yp);
    reduce_kchunks<<<512, 256, 0, stream>>>(yp, t2);
    gemm_nt_wmma_f32<<<ggrid, 128, 0, stream>>>(a1, t1, yp);
    reduce_kchunks<<<512, 256, 0, stream>>>(yp, u11);
    gemm_nt_wmma_f32<<<ggrid, 128, 0, stream>>>(a1, t2, yp);
    reduce_kchunks<<<512, 256, 0, stream>>>(yp, u12);
    gemm_nt_wmma_f32<<<ggrid, 128, 0, stream>>>(a2, t1, yp);
    reduce_kchunks<<<512, 256, 0, stream>>>(yp, u21);
    gemm_nt_wmma_f32<<<ggrid, 128, 0, stream>>>(a2, t2, yp);
    reduce_kchunks<<<512, 256, 0, stream>>>(yp, u22);

    combine_out<<<512, 256, 0, stream>>>(st, t1, t2, u11, u12, u21, u22, wemb, wout, out);
}
